// VectorQuantizer_41300405518706
// MI455X (gfx1250) — matrix operand order repaired and checked by reading, not compiled, not run on hardware
//
#include <hip/hip_runtime.h>

typedef __attribute__((ext_vector_type(16))) _Float16 v16h;
typedef __attribute__((ext_vector_type(8)))  _Float16 v8h;
typedef __attribute__((ext_vector_type(8)))  float    v8f;
typedef __attribute__((ext_vector_type(4)))  float    v4f;

#define KCODES 512
#define CDIM   64
#define HWSZ   4096          // 64*64 pixels per image plane
#define PIXWG  256           // pixels per workgroup
#define ESTR   72            // padded halves per emb row (144B, 16B-aligned, conflict-light)
#define XSTR   72            // padded halves per x-tile row

// combine two 16B LDS chunks into one 16x16-half WMMA fragment register block
static __device__ __forceinline__ v16h ld_frag(const _Float16* lo, const _Float16* hi) {
    v8h a = *reinterpret_cast<const v8h*>(lo);
    v8h b = *reinterpret_cast<const v8h*>(hi);
    v16h r;
#pragma unroll
    for (int i = 0; i < 8; ++i) { r[i] = a[i]; r[i + 8] = b[i]; }
    return r;
}

#define WMMA_F16(A, B, C) \
    __builtin_amdgcn_wmma_f32_16x16x32_f16(false, (A), false, (B), (short)0, (C), false, false)

__global__ __launch_bounds__(256)
void vq_argmin_wmma_kernel(const float* __restrict__ xg,
                           const float* __restrict__ eg,
                           float* __restrict__ out)
{
    __shared__ _Float16 emb_s[KCODES * ESTR];   // 73,728 B : codebook as f16, row-padded
    __shared__ float    esq_s[KCODES];          //  2,048 B : ||e||^2
    __shared__ _Float16 xt_s[PIXWG * XSTR];     // 36,864 B : x tile, [pixel][channel] f16
    __shared__ int      idx_s[PIXWG];           //  1,024 B : argmin per pixel

    const int t   = threadIdx.x;
    const int img = blockIdx.x >> 4;            // 16 WGs per 4096-pixel image
    const int p0  = (blockIdx.x & 15) * PIXWG;
    const float* ximg = xg + (size_t)img * CDIM * HWSZ;

    // ---- stage codebook -> LDS f16 (32768 floats = 8192 float4; 32 per thread) ----
#pragma unroll
    for (int i = 0; i < 32; ++i) {
        int f4 = i * 256 + t;
        int k  = f4 >> 4;                       // 16 float4 per 64-ch row
        int c4 = (f4 & 15) * 4;
        v4f v = *reinterpret_cast<const v4f*>(eg + k * CDIM + c4);
        _Float16* d = emb_s + k * ESTR + c4;
        d[0] = (_Float16)v.x; d[1] = (_Float16)v.y;
        d[2] = (_Float16)v.z; d[3] = (_Float16)v.w;
    }
    // ---- e_sq in fp32 (||x||^2 cancels in argmin; only e_sq - 2 x.e matters) ----
    for (int k = t; k < KCODES; k += 256) {
        const float* er = eg + k * CDIM;
        float s = 0.f;
#pragma unroll
        for (int c4 = 0; c4 < CDIM; c4 += 4) {
            v4f v = *reinterpret_cast<const v4f*>(er + c4);
            s = __builtin_fmaf(v.x, v.x, s);
            s = __builtin_fmaf(v.y, v.y, s);
            s = __builtin_fmaf(v.z, v.z, s);
            s = __builtin_fmaf(v.w, v.w, s);
        }
        esq_s[k] = s;
    }
    // ---- stage x tile transposed -> [pixel][channel] f16 (4096 float4; 16/thread) ----
#pragma unroll
    for (int i = 0; i < 16; ++i) {
        int f4 = i * 256 + t;
        int c  = f4 >> 6;                       // 64 float4 per channel row of 256 px
        int p4 = (f4 & 63) * 4;
        v4f v = *reinterpret_cast<const v4f*>(ximg + c * HWSZ + p0 + p4);
        xt_s[(p4 + 0) * XSTR + c] = (_Float16)v.x;
        xt_s[(p4 + 1) * XSTR + c] = (_Float16)v.y;
        xt_s[(p4 + 2) * XSTR + c] = (_Float16)v.z;
        xt_s[(p4 + 3) * XSTR + c] = (_Float16)v.w;
    }
    __syncthreads();

    const int lane = t & 31;
    const int wv   = t >> 5;                    // 8 waves
    const int g    = lane >> 4;                 // half-wave group (K-chunk select)
    const int lm   = lane & 15;                 // A row / B column / D column

    // A fragments for this wave's two 16-pixel tiles (contiguous K layout matching B:
    // lane holds row lm; v0..7 = K g*16..g*16+15)
    const _Float16* ar0 = xt_s + (wv * 32 + lm) * XSTR + g * 16;
    const _Float16* ar1 = ar0 + 16 * XSTR;
    const v16h a0lo = ld_frag(ar0,      ar0 + 8);    // channels 0..31
    const v16h a0hi = ld_frag(ar0 + 32, ar0 + 40);   // channels 32..63
    const v16h a1lo = ld_frag(ar1,      ar1 + 8);
    const v16h a1hi = ld_frag(ar1 + 32, ar1 + 40);

    float minv0[8], minv1[8];
    int   mini0[8], mini1[8];
#pragma unroll
    for (int r = 0; r < 8; ++r) {
        minv0[r] = 3.4e38f; minv1[r] = 3.4e38f;
        mini0[r] = 0;       mini1[r] = 0;
    }

    // ---- sweep all 512 codes, 16/tile; B fragments + e_sq software-pipelined ----
    // B layout: lane = column lm, v0..7 = K pairs; lanes 0-15 K=0..15, 16-31 K=16..31
    const _Float16* br0 = emb_s + lm * ESTR + g * 16;
    v16h  blo = ld_frag(br0,      br0 + 8);
    v16h  bhi = ld_frag(br0 + 32, br0 + 40);
    float esq = esq_s[lm];

#pragma unroll 2
    for (int cb = 0; cb < KCODES; cb += 16) {
        const int code = cb + lm;               // this lane's code column
        // prefetch next code tile (wraps to 0 on last iter; values unused then)
        const int ncb = (cb + 16) & (KCODES - 1);
        const _Float16* brn = emb_s + (ncb + lm) * ESTR + g * 16;
        const v16h  nblo = ld_frag(brn,      brn + 8);
        const v16h  nbhi = ld_frag(brn + 32, brn + 40);
        const float nesq = esq_s[ncb + lm];

        v8f acc0 = {};
        v8f acc1 = {};
        acc0 = WMMA_F16(a0lo, blo, acc0);
        acc1 = WMMA_F16(a1lo, blo, acc1);
        acc0 = WMMA_F16(a0hi, bhi, acc0);
        acc1 = WMMA_F16(a1hi, bhi, acc1);

#pragma unroll
        for (int r = 0; r < 8; ++r) {
            float d0 = __builtin_fmaf(-2.f, acc0[r], esq);   // e_sq - 2 x.e
            float d1 = __builtin_fmaf(-2.f, acc1[r], esq);
            bool  c0 = d0 < minv0[r];                        // strict < : keep first index
            bool  c1 = d1 < minv1[r];
            minv0[r] = c0 ? d0   : minv0[r];
            mini0[r] = c0 ? code : mini0[r];
            minv1[r] = c1 ? d1   : minv1[r];
            mini1[r] = c1 ? code : mini1[r];
        }

        blo = nblo; bhi = nbhi; esq = nesq;
    }

    // ---- argmin across the 16 code-lanes of each half-wave (tie -> smaller idx) ----
#pragma unroll
    for (int r = 0; r < 8; ++r) {
        float v0 = minv0[r]; int i0 = mini0[r];
        float v1 = minv1[r]; int i1 = mini1[r];
#pragma unroll
        for (int s = 8; s >= 1; s >>= 1) {
            float ov0 = __shfl_xor(v0, s, 16); int oi0 = __shfl_xor(i0, s, 16);
            float ov1 = __shfl_xor(v1, s, 16); int oi1 = __shfl_xor(i1, s, 16);
            if (ov0 < v0 || (ov0 == v0 && oi0 < i0)) { v0 = ov0; i0 = oi0; }
            if (ov1 < v1 || (ov1 == v1 && oi1 < i1)) { v1 = ov1; i1 = oi1; }
        }
        mini0[r] = i0; mini1[r] = i1;
    }
    // D layout: VGPR r <-> pixel r + g*8 within the 16-pixel tile
    if (lm == 0) {
        const int base = wv * 32;
#pragma unroll
        for (int r = 0; r < 8; ++r) idx_s[base + g * 8 + r]      = mini0[r];
#pragma unroll
        for (int r = 0; r < 8; ++r) idx_s[base + 16 + g * 8 + r] = mini1[r];
    }
    __syncthreads();

    // ---- gather fp32 codebook rows (b128 loads), write [B,C,H,W] coalesced ----
    const int pix  = t;
    const int code = idx_s[pix];
    const float* er = eg + code * CDIM;
    float* orow = out + (size_t)img * CDIM * HWSZ + p0 + pix;
#pragma unroll
    for (int c4 = 0; c4 < CDIM; c4 += 4) {
        v4f v = *reinterpret_cast<const v4f*>(er + c4);
        orow[(size_t)(c4 + 0) * HWSZ] = v.x;
        orow[(size_t)(c4 + 1) * HWSZ] = v.y;
        orow[(size_t)(c4 + 2) * HWSZ] = v.z;
        orow[(size_t)(c4 + 3) * HWSZ] = v.w;
    }
}

extern "C" void kernel_launch(void* const* d_in, const int* in_sizes, int n_in,
                              void* d_out, int out_size, void* d_ws, size_t ws_size,
                              hipStream_t stream) {
    const float* x   = (const float*)d_in[0];   // [32,64,64,64] f32
    const float* emb = (const float*)d_in[1];   // [512,64] f32
    float* out = (float*)d_out;                 // [32,64,64,64] f32
    // 512 WGs x 256 threads: each WG owns 256 pixels of one image
    vq_argmin_wmma_kernel<<<dim3(512), dim3(256), 0, stream>>>(x, emb, out);
}
